// MultiHeadAttention_9216999817434
// MI455X (gfx1250) — matrix operand order repaired and checked by reading, not compiled, not run on hardware
//
#include <hip/hip_runtime.h>
#include <hip/hip_bf16.h>

// ---------------------------------------------------------------------------
// MultiHeadAttention forward, MI455X (gfx1250), wave32 + WMMA bf16.
// B=2 S=2048 D=1024 H=16 DEPTH=64.
// Strategy: convert activations/weights to bf16 once; all GEMMs via
// v_wmma_f32_16x16x32_bf16; GEMM A-tiles staged via the CDNA5 async
// global->LDS path (ASYNCcnt); attention does one-pass softmax with the
// 16x2048 f32 score strip resident in LDS (128KB of 320KB/WGP) so the 0.54GB
// attention-weights tensor (the roofline bottleneck) is written exactly once.
// ---------------------------------------------------------------------------

typedef __attribute__((ext_vector_type(16))) __bf16       v16bf;
typedef __attribute__((ext_vector_type(8)))  float        v8f;
typedef __attribute__((ext_vector_type(4)))  unsigned int v4u;
typedef __attribute__((ext_vector_type(4)))  int          v4i;

union AF { v16bf v; v4u u[2]; };           // 32-byte WMMA bf16 fragment

#define WMMA_BF16(A, B, C) \
  __builtin_amdgcn_wmma_f32_16x16x32_bf16(false, (A), false, (B), (short)0, (C), false, false)

constexpr int BB    = 2;
constexpr int S     = 2048;
constexpr int D     = 1024;
constexpr int H     = 16;
constexpr int DEPTH = 64;
constexpr int M     = BB * S;              // 4096 token rows

// ---------------------------------------------------------------------------
// CDNA5 async global->LDS copy (per-lane 16B), tracked by ASYNCcnt.
// Builtin signature (from clang diagnostic): param0 = v4i in addrspace(1)
// (__device__), param1 = v4i in addrspace(3) (__shared__), then imm offset
// and imm cpol. (CDNA5 ISA: GLOBAL_LOAD_ASYNC_TO_LDS_B128)
// ---------------------------------------------------------------------------
typedef __attribute__((address_space(1))) v4i as1_v4i;
typedef __attribute__((address_space(3))) v4i as3_v4i;

__device__ __forceinline__ void async_copy_b128(const __bf16* g, __bf16* l) {
#if __has_builtin(__builtin_amdgcn_global_load_async_to_lds_b128)
  __builtin_amdgcn_global_load_async_to_lds_b128(
      (as1_v4i*)g, (as3_v4i*)l, 0, 0);
#else
  unsigned loff = (unsigned)(size_t)(__attribute__((address_space(3))) __bf16*)l;
  asm volatile("global_load_async_to_lds_b128 %0, %1, off"
               :: "v"(loff), "v"(g) : "memory");
#endif
}

__device__ __forceinline__ void wait_asynccnt0() {
#if __has_builtin(__builtin_amdgcn_s_wait_asynccnt)
  __builtin_amdgcn_s_wait_asynccnt(0);
#else
  asm volatile("s_wait_asynccnt 0x0" ::: "memory");
#endif
}

// ---------------------------------------------------------------------------
// Kernel 0: f32 -> bf16 convert (grid-stride)
// ---------------------------------------------------------------------------
__global__ void cvt_f32_to_bf16(const float* __restrict__ in,
                                __bf16* __restrict__ out, int n) {
  int i = blockIdx.x * blockDim.x + threadIdx.x;
  int stride = gridDim.x * blockDim.x;
  for (; i < n; i += stride) out[i] = (__bf16)in[i];
}

// ---------------------------------------------------------------------------
// Kernel 1/3: C = X(bf16, MxK row-major) @ W(bf16, NxK row-major)^T + bias
// Block = 128 threads = 4 waves. Block tile: 16 rows x 64 cols, one 16x16
// WMMA tile per wave, K-loop in steps of 64 (2 WMMAs per stage). A tile
// staged via async global->LDS (no VGPR round trip); B rows contiguous.
// MODE 0: store bf16 head-split (B,H,S,DEPTH)
// MODE 1: store bf16 transposed  (B,H,DEPTH,S)   (for V)
// MODE 2: store f32 row-major    (M,N)           (final dense output)
// ---------------------------------------------------------------------------
template <int MODE>
__global__ void __launch_bounds__(128)
gemm_xwT(const __bf16* __restrict__ X, const __bf16* __restrict__ W,
         const float* __restrict__ bias, void* __restrict__ outp, int K) {
  __shared__ __bf16 sA[16 * 64];            // 16 rows x 64 k, 2KB

  const int lane  = threadIdx.x & 31;
  const int wave  = threadIdx.x >> 5;       // 0..3
  const int l15   = lane & 15;              // A row / B col / C col
  const int khalf = lane >> 4;              // 0/1
  const int m0    = blockIdx.y * 16;
  const int n0    = blockIdx.x * 64 + wave * 16;
  const int ncol  = n0 + l15;               // this lane's output column

  // staging assignment: 1024 bf16 / 128 threads = 8 each (one b128)
  const int e  = threadIdx.x * 8;
  const int sr = e >> 6, scc = e & 63;

  const __bf16* wrow = W + (size_t)ncol * K;   // W[n,:] feeds B[k,n]
  v8f c = {};

  for (int kk = 0; kk < K; kk += 64) {
    __syncthreads();                           // previous-iter LDS reads done
    async_copy_b128(X + (size_t)(m0 + sr) * K + kk + scc, &sA[e]);
    wait_asynccnt0();                          // our async write landed
    __syncthreads();                           // all waves' stages visible

    __builtin_prefetch(wrow + kk + 64, 0, 1);  // next W chunk -> global_prefetch

#pragma unroll
    for (int ks = 0; ks < 64; ks += 32) {
      AF a, b;
      a.u[0] = *(const v4u*)(&sA[l15 * 64 + ks + khalf * 8]);
      a.u[1] = *(const v4u*)(&sA[l15 * 64 + ks + 16 + khalf * 8]);
      const __bf16* bp = wrow + kk + ks + khalf * 8;   // K vals in table order
      b.u[0] = *(const v4u*)(bp);
      b.u[1] = *(const v4u*)(bp + 16);
      c = WMMA_BF16(a.v, b.v, c);
    }
  }

  const float bv = bias[ncol];
#pragma unroll
  for (int r = 0; r < 8; ++r) {
    const int ml = r + khalf * 8;              // C layout: VGPR r, lane half
    const int m  = m0 + ml;
    const float val = c[r] + bv;
    if (MODE == 2) {
      ((float*)outp)[(size_t)m * D + ncol] = val;
    } else {
      const int b_ = m >> 11, s_ = m & (S - 1);
      const int h_ = ncol >> 6, d_ = ncol & (DEPTH - 1);
      if (MODE == 0)
        ((__bf16*)outp)[(((size_t)(b_ * H + h_) * S + s_) * DEPTH) + d_] = (__bf16)val;
      else
        ((__bf16*)outp)[(((size_t)(b_ * H + h_) * DEPTH + d_) * S) + s_] = (__bf16)val;
    }
  }
}

// ---------------------------------------------------------------------------
// Kernel 2: attention for one (b,h,16-row q tile).
// 256 threads = 8 waves. Phase 1: all 8 waves compute the 16x2048 score strip
// (QK^T / 8 + mask*-1e9) into LDS. Phase 2: one-pass softmax in LDS, write
// normalized weights (the 0.54GB output) once. Phase 3: 4 waves do P@V with
// bf16 WMMA (P converted from LDS f32 on the fly; V read from the transposed
// layout so K is contiguous), context stored bf16 for the dense projection.
// ---------------------------------------------------------------------------
__global__ void __launch_bounds__(256)
attn_block(const __bf16* __restrict__ qh, const __bf16* __restrict__ kh,
           const __bf16* __restrict__ vT, const float* __restrict__ mask,
           float* __restrict__ attw, __bf16* __restrict__ ctx) {
  extern __shared__ float smem[];
  float* sc      = smem;                  // 16 x 2048 scores / probabilities
  float* red     = smem + 16 * 2048;      // 16 x 16 partial reductions
  float* rowstat = red + 256;             // 16 row max / inv-sum

  const int lane  = threadIdx.x & 31;
  const int wave  = threadIdx.x >> 5;     // 0..7
  const int l15   = lane & 15;
  const int khalf = lane >> 4;

  const int qt = blockIdx.x & 127;        // q tile within (b,h)
  const int bh = blockIdx.x >> 7;         // 0..31
  const int b_ = bh >> 4, h_ = bh & 15;
  const int q0 = qt * 16;

  const size_t headoff = (size_t)bh * S * DEPTH;

  // ---- Phase 1: scores -----------------------------------------------------
  AF a0, a1;                              // Q fragments (d=0..31, d=32..63), reused
  {
    const __bf16* qp = qh + headoff + (size_t)(q0 + l15) * DEPTH;
    a0.u[0] = *(const v4u*)(qp + khalf * 8);
    a0.u[1] = *(const v4u*)(qp + 16 + khalf * 8);
    a1.u[0] = *(const v4u*)(qp + 32 + khalf * 8);
    a1.u[1] = *(const v4u*)(qp + 48 + khalf * 8);
  }
  for (int t = 0; t < 16; ++t) {
    const int ct   = wave * 16 + t;       // 128 column tiles split over 8 waves
    const int kcol = ct * 16 + l15;       // this lane's key column
    const __bf16* kp = kh + headoff + (size_t)kcol * DEPTH + khalf * 8;
    AF b0, b1;
    b0.u[0] = *(const v4u*)(kp);
    b0.u[1] = *(const v4u*)(kp + 16);
    b1.u[0] = *(const v4u*)(kp + 32);
    b1.u[1] = *(const v4u*)(kp + 48);
    v8f c = {};
    c = WMMA_BF16(a0.v, b0.v, c);
    c = WMMA_BF16(a1.v, b1.v, c);
    const float* mp = mask + ((size_t)b_ * S + q0) * S + kcol;  // (B,1,S,S)
#pragma unroll
    for (int r = 0; r < 8; ++r) {
      const int ml = r + khalf * 8;
      sc[ml * S + kcol] = c[r] * 0.125f + mp[(size_t)ml * S] * (-1e9f);
    }
  }
  __syncthreads();

  // ---- Phase 2: softmax + single write of attention weights ----------------
  const int srow = threadIdx.x >> 4;      // 0..15
  const int ssub = threadIdx.x & 15;      // 128-wide chunk per thread
  float* rp = sc + srow * S + ssub * 128;

  float lmax = -3.4e38f;
  for (int i = 0; i < 128; ++i) lmax = fmaxf(lmax, rp[i]);
  red[srow * 16 + ssub] = lmax;
  __syncthreads();
  if (ssub == 0) {
    float mx = red[srow * 16];
    for (int i = 1; i < 16; ++i) mx = fmaxf(mx, red[srow * 16 + i]);
    rowstat[srow] = mx;
  }
  __syncthreads();
  const float rmax = rowstat[srow];
  float lsum = 0.f;
  for (int i = 0; i < 128; ++i) { float e = __expf(rp[i] - rmax); rp[i] = e; lsum += e; }
  red[srow * 16 + ssub] = lsum;
  __syncthreads();
  if (ssub == 0) {
    float sm = 0.f;
    for (int i = 0; i < 16; ++i) sm += red[srow * 16 + i];
    rowstat[srow] = 1.0f / sm;
  }
  __syncthreads();
  const float rinv = rowstat[srow];
  float* op = attw + ((size_t)bh * S + q0 + srow) * S + ssub * 128;
  for (int i = 0; i < 128; i += 4) {
    float4 pv;
    pv.x = rp[i] * rinv; pv.y = rp[i + 1] * rinv;
    pv.z = rp[i + 2] * rinv; pv.w = rp[i + 3] * rinv;
    rp[i] = pv.x; rp[i + 1] = pv.y; rp[i + 2] = pv.z; rp[i + 3] = pv.w;
    *(float4*)(op + i) = pv;              // the only touch of the 0.5GB tensor
  }
  __syncthreads();

  // ---- Phase 3: P @ V (4 waves, one 16x16 d-tile each) ---------------------
  if (wave < 4) {
    const int dt = wave;
    const __bf16* vp = vT + (size_t)bh * DEPTH * S
                          + (size_t)(dt * 16 + l15) * S + khalf * 8;
    v8f c = {};
    for (int kk = 0; kk < S; kk += 32) {
      union F8 { float4 f[2]; float s[8]; };
      F8 lo, hi;
      const float* p0 = sc + l15 * S + kk + khalf * 8;
      lo.f[0] = *(const float4*)(p0);      lo.f[1] = *(const float4*)(p0 + 4);
      hi.f[0] = *(const float4*)(p0 + 16); hi.f[1] = *(const float4*)(p0 + 20);
      AF a, bb;
#pragma unroll
      for (int i = 0; i < 8; ++i) { a.v[i] = (__bf16)lo.s[i]; a.v[8 + i] = (__bf16)hi.s[i]; }
      bb.u[0] = *(const v4u*)(vp + kk);
      bb.u[1] = *(const v4u*)(vp + kk + 16);
      c = WMMA_BF16(a.v, bb.v, c);
    }
#pragma unroll
    for (int r = 0; r < 8; ++r) {
      const int ml = r + khalf * 8;
      const int q  = q0 + ml;
      const int dc = dt * 16 + l15;
      ctx[((size_t)b_ * S + q) * D + h_ * DEPTH + dc] = (__bf16)c[r];
    }
  }
}

// ---------------------------------------------------------------------------
// Host orchestration
// ---------------------------------------------------------------------------
extern "C" void kernel_launch(void* const* d_in, const int* in_sizes, int n_in,
                              void* d_out, int out_size, void* d_ws, size_t ws_size,
                              hipStream_t stream) {
  // input order: q, v, k, mask, wq_w, wq_b, wk_w, wk_b, wv_w, wv_b, dense_w, dense_b
  const float* q_in  = (const float*)d_in[0];
  const float* v_in  = (const float*)d_in[1];
  const float* k_in  = (const float*)d_in[2];
  const float* mask  = (const float*)d_in[3];
  const float* wq_w  = (const float*)d_in[4];
  const float* wq_b  = (const float*)d_in[5];
  const float* wk_w  = (const float*)d_in[6];
  const float* wk_b  = (const float*)d_in[7];
  const float* wv_w  = (const float*)d_in[8];
  const float* wv_b  = (const float*)d_in[9];
  const float* wd_w  = (const float*)d_in[10];
  const float* wd_b  = (const float*)d_in[11];

  float* out_dense = (float*)d_out;                       // (B,S,D)
  float* out_attw  = (float*)d_out + (size_t)M * D;       // (B,H,S,S)

  // workspace partition (bf16 elements), total 32M elems = 64MB
  __bf16* ws  = (__bf16*)d_ws;
  __bf16* qb  = ws;                       // 4M  activations bf16
  __bf16* vb  = qb  + (size_t)M * D;
  __bf16* kb  = vb  + (size_t)M * D;
  __bf16* wqb = kb  + (size_t)M * D;      // 1M weights bf16
  __bf16* wkb = wqb + (size_t)D * D;
  __bf16* wvb = wkb + (size_t)D * D;
  __bf16* wdb = wvb + (size_t)D * D;
  __bf16* qhb = wdb + (size_t)D * D;      // 4M head-split projections
  __bf16* khb = qhb + (size_t)M * D;
  __bf16* vTb = khb + (size_t)M * D;      // transposed V (B,H,DEPTH,S)
  __bf16* ctx = vTb + (size_t)M * D;      // attention context bf16

  // 0) convert everything that feeds WMMA to bf16, once
  cvt_f32_to_bf16<<<2048, 256, 0, stream>>>(q_in, qb, M * D);
  cvt_f32_to_bf16<<<2048, 256, 0, stream>>>(v_in, vb, M * D);
  cvt_f32_to_bf16<<<2048, 256, 0, stream>>>(k_in, kb, M * D);
  cvt_f32_to_bf16<<<1024, 256, 0, stream>>>(wq_w, wqb, D * D);
  cvt_f32_to_bf16<<<1024, 256, 0, stream>>>(wk_w, wkb, D * D);
  cvt_f32_to_bf16<<<1024, 256, 0, stream>>>(wv_w, wvb, D * D);
  cvt_f32_to_bf16<<<1024, 256, 0, stream>>>(wd_w, wdb, D * D);

  // 1) QKV projections (WMMA bf16), V stored transposed for the PV matmul
  dim3 ggrid(D / 64, M / 16);
  gemm_xwT<0><<<ggrid, 128, 0, stream>>>(qb, wqb, wq_b, qhb, D);
  gemm_xwT<0><<<ggrid, 128, 0, stream>>>(kb, wkb, wk_b, khb, D);
  gemm_xwT<1><<<ggrid, 128, 0, stream>>>(vb, wvb, wv_b, vTb, D);

  // 2) attention: 16x2048 score strip in LDS (128KB of 320KB/WGP)
  const size_t smem_bytes = (16 * 2048 + 256 + 16) * sizeof(float);  // ~129KB
  (void)hipFuncSetAttribute((const void*)attn_block,
                            hipFuncAttributeMaxDynamicSharedMemorySize,
                            (int)smem_bytes);
  attn_block<<<BB * H * (S / 16), 256, smem_bytes, stream>>>(
      qhb, khb, vTb, mask, out_attw, ctx);

  // 3) dense output projection -> f32 d_out
  gemm_xwT<2><<<ggrid, 128, 0, stream>>>(ctx, wdb, wd_b, out_dense, D);
}
